// SegAttention_75127567941747
// MI455X (gfx1250) — matrix operand order repaired and checked by reading, not compiled, not run on hardware
//
#include <hip/hip_runtime.h>
#include <hip/hip_bf16.h>
#include <math.h>

// Problem constants (from reference)
#define SQ   256   // S: sequence length
#define NB   1024  // N: batch
#define HD   512   // H: hidden
#define AD   128   // attention dim
#define ROWE 520   // padded LDS row length in bf16 elems (512 + 8 pad -> 1040B, 16B aligned, banks offset by 4)

typedef __attribute__((ext_vector_type(16))) __bf16 v16bf;
typedef __attribute__((ext_vector_type(8)))  __bf16 v8bf;
typedef __attribute__((ext_vector_type(4)))  __bf16 v4bf;
typedef __attribute__((ext_vector_type(2)))  __bf16 v2bf;
typedef __attribute__((ext_vector_type(8)))  float  v8f;
typedef __attribute__((ext_vector_type(4)))  float  v4f;

// One workgroup (256 threads = 8 wave32) handles one n of one branch.
// LDS: x as bf16 [256][520] (260KB) + scores f32 [256][2] (2KB).
__global__ void __launch_bounds__(256)
seg_attn_pool_kernel(const float* __restrict__ emb,          // [S, N, H]
                     const unsigned char* __restrict__ mask, // [N, S] bool
                     const float* __restrict__ W1,           // [H, AD]
                     const float* __restrict__ W2,           // [AD, 2]
                     float* __restrict__ pooled)             // [N, H] mean over the 2 heads
{
    extern __shared__ char smem_raw[];
    __bf16* xs     = (__bf16*)smem_raw;                          // [SQ][ROWE]
    float*  scores = (float*)(smem_raw + (size_t)SQ * ROWE * 2); // [SQ][2]

    const int tid  = threadIdx.x;
    const int lane = tid & 31;
    const int wave = tid >> 5;
    const int n    = blockIdx.x;

    // ---- zero score accumulator ----
    if (tid < SQ * 2) scores[tid] = 0.0f;

    // ---- Phase 1: stream x = emb[:, n, :] from HBM once, convert f32->bf16 into LDS ----
    // wave w takes rows s = w, w+8, ...; each row (2KB) loaded as 4 coalesced 512B wave-loads.
    for (int s = wave; s < SQ; s += 8) {
        const float* src = emb + ((size_t)s * NB + n) * HD;
        __bf16* dst = xs + s * ROWE;
        __builtin_prefetch(emb + ((size_t)(s + 8) * NB + n) * HD, 0, 3); // global_prefetch_b8
#pragma unroll
        for (int it = 0; it < 4; ++it) {
            v4f f = *(const v4f*)(src + it * 128 + lane * 4);   // global_load_b128
            v4bf b;
            b[0] = (__bf16)f.x; b[1] = (__bf16)f.y;
            b[2] = (__bf16)f.z; b[3] = (__bf16)f.w;
            *(v4bf*)(dst + it * 128 + lane * 4) = b;            // ds_store_b64
        }
    }
    __syncthreads();

    // ---- Phase 2: scores = tanh(x @ W1) @ W2 via bf16 WMMA ----
    // Wave w owns AD columns [w*16, w*16+16). B panel (W1 columns, bf16) resident in VGPRs.
    const int col   = lane & 15;   // N-column within tile / C-layout column
    const int khalf = lane >> 4;   // which 16-lane half
    const int ad0   = wave * 16;

    // B layout used here (contiguous k, matched by A): lanes 0-15 hold K=0..15 of column `col`
    // (2 packed per VGPR across 8 VGPRs), lanes 16-31 hold K=16..31.
    v16bf Bp[16];
    {
        const float* w1c = W1 + (ad0 + col);   // W1[k][ad0+col] = w1c[k*AD]
#pragma unroll
        for (int kt = 0; kt < 16; ++kt) {
            const int kb = kt * 32 + khalf * 16;
#pragma unroll
            for (int v = 0; v < 8; ++v) {
                float f0 = w1c[(size_t)(kb + 2 * v) * AD];
                float f1 = w1c[(size_t)(kb + 2 * v + 1) * AD];
                Bp[kt][2 * v]     = (__bf16)f0;
                Bp[kt][2 * v + 1] = (__bf16)f1;
            }
        }
    }
    const float w2r0 = W2[(ad0 + col) * 2 + 0];
    const float w2r1 = W2[(ad0 + col) * 2 + 1];

    for (int st = 0; st < 16; ++st) {
        v8f acc = {};
        // A layout (matches B): lane half 0 holds K 0..15, half 1 holds K 16..31
        const __bf16* rowp = xs + (size_t)(st * 16 + col) * ROWE;

        // One-stage software pipeline: issue LDS loads for tile kt+1 before the
        // WMMA consuming tile kt, so ds latency overlaps the matrix pipe.
        v8bf alo = *(const v8bf*)(rowp + khalf * 16);       // ds_load_b128
        v8bf ahi = *(const v8bf*)(rowp + khalf * 16 + 8);   // ds_load_b128
#pragma unroll
        for (int kt = 0; kt < 16; ++kt) {
            v8bf nlo = {}, nhi = {};
            if (kt < 15) {
                const int noff = (kt + 1) * 32 + khalf * 16;
                nlo = *(const v8bf*)(rowp + noff);
                nhi = *(const v8bf*)(rowp + noff + 8);
            }
            v16bf a;
#pragma unroll
            for (int e = 0; e < 8; ++e) { a[e] = alo[e]; a[e + 8] = ahi[e]; }
            acc = __builtin_amdgcn_wmma_f32_16x16x32_bf16(
                false, a, false, Bp[kt], (short)0, acc, false, false);
            alo = nlo; ahi = nhi;
        }
        // C layout: VGPR j, lanes 0-15 -> M=j, lanes 16-31 -> M=j+8; N = col.
        float q0[8], q1[8];
#pragma unroll
        for (int j = 0; j < 8; ++j) {
            float tv = tanhf(acc[j]);
            q0[j] = tv * w2r0;
            q1[j] = tv * w2r1;
        }
        // reduce over the 16 AD columns held by this lane-half (wave32 butterflies)
#pragma unroll
        for (int m = 1; m < 16; m <<= 1) {
#pragma unroll
            for (int j = 0; j < 8; ++j) {
                q0[j] += __shfl_xor(q0[j], m, 32);
                q1[j] += __shfl_xor(q1[j], m, 32);
            }
        }
        if (col == 0) {
#pragma unroll
            for (int j = 0; j < 8; ++j) {
                const int sg = st * 16 + khalf * 8 + j;
                atomicAdd(&scores[sg * 2 + 0], q0[j]);   // ds_add_f32
                atomicAdd(&scores[sg * 2 + 1], q1[j]);
            }
        }
    }
    __syncthreads();

    // ---- Phase 3: softmax over S, mask(-inf), softmax over S again (waves 0 and 1, r = wave) ----
    if (wave < 2) {
        const int r = wave;
        float v[8];
#pragma unroll
        for (int i = 0; i < 8; ++i) v[i] = scores[(i * 32 + lane) * 2 + r];

        float mx = v[0];
#pragma unroll
        for (int i = 1; i < 8; ++i) mx = fmaxf(mx, v[i]);
#pragma unroll
        for (int m = 1; m < 32; m <<= 1) mx = fmaxf(mx, __shfl_xor(mx, m, 32));
        float sum = 0.0f;
#pragma unroll
        for (int i = 0; i < 8; ++i) { v[i] = expf(v[i] - mx); sum += v[i]; }
#pragma unroll
        for (int m = 1; m < 32; m <<= 1) sum += __shfl_xor(sum, m, 32);
        const float inv = 1.0f / sum;

#pragma unroll
        for (int i = 0; i < 8; ++i) {
            float p = v[i] * inv;
            bool mk = mask[(size_t)n * SQ + i * 32 + lane] != 0;
            v[i] = mk ? -INFINITY : p;
        }
        float mx2 = v[0];
#pragma unroll
        for (int i = 1; i < 8; ++i) mx2 = fmaxf(mx2, v[i]);
#pragma unroll
        for (int m = 1; m < 32; m <<= 1) mx2 = fmaxf(mx2, __shfl_xor(mx2, m, 32));
        float sum2 = 0.0f;
#pragma unroll
        for (int i = 0; i < 8; ++i) { v[i] = expf(v[i] - mx2); sum2 += v[i]; }
#pragma unroll
        for (int m = 1; m < 32; m <<= 1) sum2 += __shfl_xor(sum2, m, 32);
        const float inv2 = 1.0f / sum2;
#pragma unroll
        for (int i = 0; i < 8; ++i) scores[(i * 32 + lane) * 2 + r] = v[i] * inv2;
    }
    __syncthreads();

    // ---- Phase 4: pooled[n][h] = 0.5 * sum_r sum_s w[s][r] * x[s][h] (x from LDS, f32 accum) ----
    const int h0 = tid * 2;  // 256 threads x 2 h each = 512
    float a0 = 0.f, a1 = 0.f, b0 = 0.f, b1 = 0.f;
#pragma unroll 4
    for (int s = 0; s < SQ; ++s) {
        const float w0 = scores[s * 2 + 0];
        const float w1 = scores[s * 2 + 1];
        v2bf xv = *(const v2bf*)(xs + (size_t)s * ROWE + h0);  // ds_load_b32
        const float x0 = (float)xv[0];
        const float x1 = (float)xv[1];
        a0 += w0 * x0; b0 += w1 * x0;
        a1 += w0 * x1; b1 += w1 * x1;
    }
    pooled[(size_t)n * HD + h0]     = 0.5f * (a0 + b0);
    pooled[(size_t)n * HD + h0 + 1] = 0.5f * (a1 + b1);
}

// pos[i] = cumsum(valid)[i] - 1  (tiny serial scan; N=1024)
__global__ void seg_attn_scan_kernel(const int* __restrict__ valid, int* __restrict__ pos, int nn)
{
    if (threadIdx.x == 0 && blockIdx.x == 0) {
        int c = 0;
        for (int i = 0; i < nn; ++i) { c += (valid[i] != 0); pos[i] = c - 1; }
    }
}

// out[n, 0, br*H + h] = valid[n] ? pooled_br[pos[n]][h] : 0
__global__ void seg_attn_gather_kernel(const float* __restrict__ pooled0,
                                       const float* __restrict__ pooled1,
                                       const int* __restrict__ valid0,
                                       const int* __restrict__ valid1,
                                       const int* __restrict__ pos0,
                                       const int* __restrict__ pos1,
                                       float* __restrict__ out)
{
    const int idx = blockIdx.x * blockDim.x + threadIdx.x; // N * 2H
    const int n  = idx >> 10;
    const int f  = idx & 1023;
    const int br = f >> 9;
    const int h  = f & 511;
    const float* pooled = br ? pooled1 : pooled0;
    const int*   valid  = br ? valid1  : valid0;
    const int*   pos    = br ? pos1    : pos0;
    float r = 0.0f;
    if (valid[n] != 0) r = pooled[(size_t)pos[n] * HD + h];
    out[idx] = r;
}

extern "C" void kernel_launch(void* const* d_in, const int* in_sizes, int n_in,
                              void* d_out, int out_size, void* d_ws, size_t ws_size,
                              hipStream_t stream) {
    (void)in_sizes; (void)n_in; (void)out_size; (void)ws_size;
    const float*         pre_emb   = (const float*)d_in[0];
    const float*         fol_emb   = (const float*)d_in[1];
    const unsigned char* pre_mask  = (const unsigned char*)d_in[2];
    const unsigned char* fol_mask  = (const unsigned char*)d_in[3];
    const int*           pre_valid = (const int*)d_in[4];
    const int*           fol_valid = (const int*)d_in[5];
    const float*         W1_pre    = (const float*)d_in[6];
    const float*         W2_pre    = (const float*)d_in[7];
    const float*         W1_fol    = (const float*)d_in[8];
    const float*         W2_fol    = (const float*)d_in[9];
    float* out = (float*)d_out;

    char* ws = (char*)d_ws;
    int*   pos0    = (int*)ws;                                   // 4KB
    int*   pos1    = (int*)(ws + 4096);                          // 4KB
    float* pooled0 = (float*)(ws + 8192);                        // 2MB
    float* pooled1 = (float*)(ws + 8192 + (size_t)NB * HD * 4);  // 2MB

    const size_t smem = (size_t)SQ * ROWE * 2 + (size_t)SQ * 2 * 4; // 268,288 B (< 320KB WGP LDS)
    hipFuncSetAttribute((const void*)seg_attn_pool_kernel,
                        hipFuncAttributeMaxDynamicSharedMemorySize, (int)smem);

    seg_attn_pool_kernel<<<NB, 256, smem, stream>>>(pre_emb, pre_mask, W1_pre, W2_pre, pooled0);
    seg_attn_pool_kernel<<<NB, 256, smem, stream>>>(fol_emb, fol_mask, W1_fol, W2_fol, pooled1);
    seg_attn_scan_kernel<<<1, 32, 0, stream>>>(pre_valid, pos0, NB);
    seg_attn_scan_kernel<<<1, 32, 0, stream>>>(fol_valid, pos1, NB);
    seg_attn_gather_kernel<<<(NB * 2 * HD) / 256, 256, 0, stream>>>(
        pooled0, pooled1, pre_valid, fol_valid, pos0, pos1, out);
}
